// CosineSimCodebook_61108794688006
// MI455X (gfx1250) — matrix operand order repaired and checked by reading, not compiled, not run on hardware
//
#include <hip/hip_runtime.h>

typedef __attribute__((ext_vector_type(16))) __bf16 v16bf;
typedef __attribute__((ext_vector_type(8)))  __bf16 v8bf;
typedef __attribute__((ext_vector_type(8)))  float  v8f;
typedef __attribute__((ext_vector_type(4)))  int    v4i;

#define DDIM    512
#define NCODES  8192
#define NROWS   32768
#define MTILE   128
#define NCHUNK  64
#define NCHUNKS (NCODES / NCHUNK)
#define THREADS 256
#define RED_BYTES (2 * 128 * 4 + 2 * 128 * 4 + 128 * 4)  // redV + redI + bestIdx

#if defined(__gfx1250__) && __has_builtin(__builtin_amdgcn_global_load_async_to_lds_b128)
#define ASYNC_OK 1
#else
#define ASYNC_OK 0
#endif

#define AS1 __attribute__((address_space(1)))
#define AS3 __attribute__((address_space(3)))

__device__ __forceinline__ unsigned short f2bf(float f) {
    unsigned int u = __float_as_uint(f);
    u += 0x7FFFu + ((u >> 16) & 1u);          // round-to-nearest-even
    return (unsigned short)(u >> 16);
}

// ---------------------------------------------------------------------------
// Prep kernel: invnorm[c] = 1/max(||embed_c||,1e-12); optionally also emit the
// normalized bf16 codebook into workspace (bcode != nullptr).
// ---------------------------------------------------------------------------
__global__ __launch_bounds__(128)
void prep_kernel(const float* __restrict__ embed,
                 float* __restrict__ invnorm,
                 unsigned short* __restrict__ bcode) {
    const int row = blockIdx.x;
    const float* e = embed + (size_t)row * DDIM;
    float s = 0.0f;
    for (int i = threadIdx.x; i < DDIM; i += 128) {
        float v = e[i];
        s += v * v;
    }
    for (int m = 16; m >= 1; m >>= 1) s += __shfl_xor(s, m);
    __shared__ float wsum[4];
    __shared__ float sInv;
    const int lane = threadIdx.x & 31;
    const int wv   = threadIdx.x >> 5;
    if (lane == 0) wsum[wv] = s;
    __syncthreads();
    if (threadIdx.x == 0) {
        float t = wsum[0] + wsum[1] + wsum[2] + wsum[3];
        float inv = 1.0f / fmaxf(sqrtf(t), 1e-12f);
        invnorm[row] = inv;
        sInv = inv;
    }
    __syncthreads();
    if (bcode) {
        const float inv = sInv;
        unsigned short* bb = bcode + (size_t)row * DDIM;
        for (int i = threadIdx.x; i < DDIM; i += 128)
            bb[i] = f2bf(e[i] * inv);
    }
}

// ---------------------------------------------------------------------------
// Async stage: copy one 64x512 bf16 chunk (64KB) global -> LDS, ASYNCcnt-tracked
// (16 global_load_async_to_lds_b128 per thread)
// ---------------------------------------------------------------------------
#if ASYNC_OK
__device__ __forceinline__ void stage_async(const unsigned short* src,
                                            unsigned short* dst) {
    const int tid = threadIdx.x;
#pragma unroll
    for (int i = 0; i < 16; ++i) {
        const int off = (tid + i * THREADS) * 16;   // bytes
        __builtin_amdgcn_global_load_async_to_lds_b128(
            (AS1 v4i*)((const char*)src + off),
            (AS3 v4i*)((char*)dst + off),
            0, 0);
    }
}
#endif

// ---------------------------------------------------------------------------
// Main kernel: fused bf16-WMMA GEMM + running argmax + codebook gather.
//   grid.x = NROWS/MTILE, 256 threads = 8 waves: mg = wave>>1 (4 row groups of
//   32), ng = wave&1 (2 col groups of 32). Each wave: 2x2 WMMA fragments.
//   BF16B=true : B chunks async-double-buffered from bf16 codebook in ws.
//   BF16B=false: B chunks converted f32->bf16 in-loop (single buffer).
// ---------------------------------------------------------------------------
template <bool BF16B>
__global__ __launch_bounds__(THREADS)
void vq_argmax_gather_kernel(const float* __restrict__ x,
                             const float* __restrict__ embed,
                             const float* __restrict__ invnorm,
                             const unsigned short* __restrict__ bcode,
                             float* __restrict__ outq,
                             int* __restrict__ outidx) {
    extern __shared__ unsigned short smem[];
    unsigned short* Albf = smem;                               // MTILE*DDIM
    unsigned short* B0   = smem + MTILE * DDIM;                // NCHUNK*DDIM
    unsigned short* B1   = BF16B ? (B0 + NCHUNK * DDIM) : B0;  // dbl buffer
    unsigned short* redB = BF16B ? (B1 + NCHUNK * DDIM) : (B0 + NCHUNK * DDIM);
    float* redV   = (float*)redB;     // [2][128]
    int*   redI   = (int*)(redV + 256);
    int*   bestIx = redI + 256;       // [128]

    const int tid  = threadIdx.x;
    const int wave = tid >> 5;
    const int lane = tid & 31;
    const int l16  = lane & 15;
    const int hi   = lane >> 4;           // lane half (C-matrix row split)
    const int mg   = wave >> 1;           // row group: rows [mg*32, mg*32+32)
    const int ng   = wave & 1;            // col group: cols [ng*32, ng*32+32)

    const size_t rowBase = (size_t)blockIdx.x * MTILE;

    // ---- stage A tile once: f32 global -> bf16 LDS (row-normalization of x
    //      is a per-row positive scale -> argmax-invariant, skip it) ----
    {
        const float4* src = (const float4*)(x + rowBase * DDIM);
        for (int i = tid; i < MTILE * DDIM / 4; i += THREADS) {
            float4 v = src[i];
            unsigned short* dst = Albf + i * 4;
            dst[0] = f2bf(v.x); dst[1] = f2bf(v.y);
            dst[2] = f2bf(v.z); dst[3] = f2bf(v.w);
        }
    }

    float mval[2][8];
    int   midx[2][8];
#pragma unroll
    for (int i = 0; i < 2; ++i)
#pragma unroll
        for (int s = 0; s < 8; ++s) { mval[i][s] = -3.402823466e38f; midx[i][s] = 0; }

#if ASYNC_OK
    if (BF16B) stage_async(bcode, B0);   // prologue: chunk 0 -> B0
#endif

    for (int chunk = 0; chunk < NCHUNKS; ++chunk) {
        unsigned short* Bcur;
        if (BF16B) {
#if ASYNC_OK
            Bcur = (chunk & 1) ? B1 : B0;
            if (chunk + 1 < NCHUNKS) {
                stage_async(bcode + (size_t)(chunk + 1) * NCHUNK * DDIM,
                            (chunk & 1) ? B0 : B1);
                asm volatile("s_wait_asynccnt 0x10" ::: "memory");
            } else {
                asm volatile("s_wait_asynccnt 0x0" ::: "memory");
            }
            __syncthreads();
#else
            Bcur = B0;
            __syncthreads();
            {   // plain bf16 chunk copy
                const uint4* s4 = (const uint4*)(bcode + (size_t)chunk * NCHUNK * DDIM);
                uint4* d4 = (uint4*)B0;
                for (int i = tid; i < NCHUNK * DDIM / 8; i += THREADS) d4[i] = s4[i];
            }
            __syncthreads();
#endif
        } else {
            Bcur = B0;
            __syncthreads();
            {   // f32 chunk -> scaled bf16 LDS
                const float4* src = (const float4*)(embed + (size_t)chunk * NCHUNK * DDIM);
                for (int i = tid; i < NCHUNK * DDIM / 4; i += THREADS) {
                    float4 v = src[i];
                    const int r = i / (DDIM / 4);
                    const float sc = invnorm[chunk * NCHUNK + r];
                    unsigned short* dst = B0 + i * 4;
                    dst[0] = f2bf(v.x * sc); dst[1] = f2bf(v.y * sc);
                    dst[2] = f2bf(v.z * sc); dst[3] = f2bf(v.w * sc);
                }
            }
            if (chunk + 1 < NCHUNKS) {
                const char* nxt = (const char*)(embed + (size_t)(chunk + 1) * NCHUNK * DDIM);
                for (int i = tid; i < (NCHUNK * DDIM * 4) / 128; i += THREADS)
                    __builtin_prefetch(nxt + i * 128, 0, 1);
            }
            __syncthreads();
        }

        v8f acc[2][2];
#pragma unroll
        for (int i = 0; i < 2; ++i)
#pragma unroll
            for (int j = 0; j < 2; ++j) {
                v8f z = {0.f, 0.f, 0.f, 0.f, 0.f, 0.f, 0.f, 0.f};
                acc[i][j] = z;
            }

#pragma unroll 4
        for (int k0 = 0; k0 < DDIM; k0 += 32) {
            // A fragments (16x32 bf16): lanes 0-15 -> K 0-15,
            //                           lanes 16-31 -> K 16-31
            union { v16bf v; v8bf h[2]; } a[2];
#pragma unroll
            for (int i = 0; i < 2; ++i) {
                const unsigned short* ap =
                    Albf + (mg * 32 + i * 16 + l16) * DDIM + k0 + hi * 16;
                a[i].h[0] = *(const v8bf*)(ap);
                a[i].h[1] = *(const v8bf*)(ap + 8);
            }
            // B fragments (32x16 bf16): lane n holds K 0-15 / 16-31, 32B each
            v16bf b[2];
#pragma unroll
            for (int j = 0; j < 2; ++j) {
                const unsigned short* bp =
                    Bcur + (ng * 32 + j * 16 + l16) * DDIM + k0 + hi * 16;
                b[j] = *(const v16bf*)bp;
            }
#pragma unroll
            for (int i = 0; i < 2; ++i)
#pragma unroll
                for (int j = 0; j < 2; ++j)
                    acc[i][j] = __builtin_amdgcn_wmma_f32_16x16x32_bf16(
                        false, a[i].v, false, b[j], (short)0, acc[i][j],
                        false, false);
        }

        // ---- running argmax update (lane = column, slot s = row) ----
#pragma unroll
        for (int j = 0; j < 2; ++j) {
            const int c = chunk * NCHUNK + ng * 32 + j * 16 + l16;
#pragma unroll
            for (int i = 0; i < 2; ++i)
#pragma unroll
                for (int s = 0; s < 8; ++s) {
                    float v = acc[i][j][s];
                    if (v > mval[i][s]) { mval[i][s] = v; midx[i][s] = c; }
                }
        }
        __syncthreads();   // all waves done reading Bcur before it is restaged
    }

    // ---- reduce across the 16 lanes holding each row (masks stay in-half) ----
#pragma unroll
    for (int i = 0; i < 2; ++i)
#pragma unroll
        for (int s = 0; s < 8; ++s) {
#pragma unroll
            for (int m = 1; m < 16; m <<= 1) {
                float ov = __shfl_xor(mval[i][s], m);
                int   oi = __shfl_xor(midx[i][s], m);
                if (ov > mval[i][s] || (ov == mval[i][s] && oi < midx[i][s])) {
                    mval[i][s] = ov; midx[i][s] = oi;
                }
            }
        }
    if (l16 == 0) {
        // lane 0 holds rows +0..7, lane 16 holds rows +8..15 of each fragment
#pragma unroll
        for (int i = 0; i < 2; ++i)
#pragma unroll
            for (int s = 0; s < 8; ++s) {
                const int r = mg * 32 + i * 16 + hi * 8 + s;
                redV[ng * 128 + r] = mval[i][s];
                redI[ng * 128 + r] = midx[i][s];
            }
    }
    __syncthreads();

    if (tid < MTILE) {
        float v0 = redV[tid];       int i0 = redI[tid];
        float v1 = redV[128 + tid]; int i1 = redI[128 + tid];
        int best = (v1 > v0 || (v1 == v0 && i1 < i0)) ? i1 : i0;
        bestIx[tid] = best;
        outidx[rowBase + tid] = best;
    }
    __syncthreads();

    // ---- gather un-normalized codebook rows ----
    const float4* e4 = (const float4*)embed;
    float4* o4 = (float4*)(outq + rowBase * DDIM);
    for (int i = tid; i < MTILE * (DDIM / 4); i += THREADS) {
        const int r = i >> 7;            // DDIM/4 == 128
        o4[i] = e4[(size_t)bestIx[r] * (DDIM / 4) + (i & 127)];
    }
}

// ---------------------------------------------------------------------------
extern "C" void kernel_launch(void* const* d_in, const int* in_sizes, int n_in,
                              void* d_out, int out_size, void* d_ws, size_t ws_size,
                              hipStream_t stream) {
    const float* x     = (const float*)d_in[0];   // [16,2048,512] f32
    const float* embed = (const float*)d_in[1];   // [1,8192,512] f32
    float* outq   = (float*)d_out;                       // 16*2048*512 f32
    int*   outidx = (int*)d_out + (size_t)NROWS * DDIM;  // 32768 i32

    float* invnorm = (float*)d_ws;                                   // 32KB
    unsigned short* bcode = (unsigned short*)((char*)d_ws + NCODES * sizeof(float));
    const size_t wsNeeded = NCODES * sizeof(float)
                          + (size_t)NCODES * DDIM * sizeof(unsigned short);
    const bool useBf = (ws_size >= wsNeeded);

    hipLaunchKernelGGL(prep_kernel, dim3(NCODES), dim3(128), 0, stream,
                       embed, invnorm, useBf ? bcode : (unsigned short*)nullptr);

    if (useBf) {
        const size_t lds = (size_t)(MTILE + 2 * NCHUNK) * DDIM * 2 + RED_BYTES;
        hipLaunchKernelGGL((vq_argmax_gather_kernel<true>),
                           dim3(NROWS / MTILE), dim3(THREADS), lds, stream,
                           x, embed, invnorm, bcode, outq, outidx);
    } else {
        const size_t lds = (size_t)(MTILE + NCHUNK) * DDIM * 2 + RED_BYTES;
        hipLaunchKernelGGL((vq_argmax_gather_kernel<false>),
                           dim3(NROWS / MTILE), dim3(THREADS), lds, stream,
                           x, embed, invnorm, (const unsigned short*)nullptr,
                           outq, outidx);
    }
}
